// _SiameseCrossAttentionPolicy_2954937499686
// MI455X (gfx1250) — matrix operand order repaired and checked by reading, not compiled, not run on hardware
//
#include <hip/hip_runtime.h>
#include <hip/hip_bf16.h>

typedef __attribute__((ext_vector_type(16))) _Float16 v16h;
typedef __attribute__((ext_vector_type(8)))  _Float16 v8h;
typedef __attribute__((ext_vector_type(8)))  float    v8f;
typedef __attribute__((ext_vector_type(4)))  float    v4f;

#define DEVINL __device__ __forceinline__

// LDS leading dims padded by 8 halfs (16 B) so row stride is not a multiple of
// the 256 B bank period: 16 row-parallel v8h loads spread over all 64 banks.
#define MLD  1032   // merged buffer ld (1024 cols + 8 pad)
#define HLD  520    // hidden buffer ld (512 + 8)
#define H2LD 264    // merge2 out ld (256 + 8)
#define PLD  40     // q/k/v token buffer ld (32 + 8)

#define MERGED_HALFS (16 * MLD)          // 16512
#define SCR_HALFS    (3 * 128 * PLD)     // 15360 (covers qkv; also hbuf+h2)
#define BLK_HALFS    (MERGED_HALFS + SCR_HALFS)   // 31872 halfs = 62.25 KB
#define NWAVES 4

// ---------------------------------------------------------------------------
// WMMA helpers (CDNA5 wave32, 16x16x32 f16 -> f32)
// ---------------------------------------------------------------------------
DEVINL v8f wmma_f16(v16h a, v16h b, v8f c) {
  return __builtin_amdgcn_wmma_f32_16x16x32_f16(
      /*neg_a=*/false, a, /*neg_b=*/false, b,
      /*c_mod=*/(short)0, c, /*reuse_a=*/false, /*reuse_b=*/false);
}

// B fragments are pre-swizzled in workspace: frag = nt*Ktiles + kt,
// lane l owns 16 contiguous halfs at ws[frag*512 + l*16].
DEVINL v16h load_b_frag(const _Float16* ws, int frag, int lane) {
  return *(const v16h*)(ws + (size_t)frag * 512 + (size_t)lane * 16);
}

// Lane-spread prefetch of the next fragment chunk (emits global_prefetch_b8).
DEVINL void prefetch_chunk(const _Float16* p, int lane, int laneStrideHalfs) {
  __builtin_prefetch((const void*)(p + (size_t)lane * laneStrideHalfs), 0, 1);
}

// A fragment (16x32 f16) from a row-major f16 buffer (LDS), leading dim ldA.
// ISA layout: lanes 0-15 -> M=lane, elems 0..7 = K k0..k0+7, 8..15 = K k0+16..k0+23
//             lanes 16-31 -> M=lane-16, K shifted by +8.
DEVINL v16h load_a_frag_lds(const _Float16* buf, int ldA, int k0, int lane) {
  int m  = lane & 15;
  int ks = (lane >> 4) * 8;
  const _Float16* p = buf + (size_t)m * ldA + k0 + ks;
  v8h lo = *(const v8h*)(p);
  v8h hi = *(const v8h*)(p + 16);
  v16h a;
#pragma unroll
  for (int i = 0; i < 8; ++i) { a[i] = lo[i]; a[i + 8] = hi[i]; }
  return a;
}

// A fragment loaded directly from global f32 (converted to f16).
DEVINL v16h load_a_frag_global_f32(const float* g, int ldg, int k0, int lane) {
  int m  = lane & 15;
  int ks = (lane >> 4) * 8;
  const float* p = g + (size_t)m * ldg + k0 + ks;
  v4f f0 = *(const v4f*)(p);
  v4f f1 = *(const v4f*)(p + 4);
  v4f f2 = *(const v4f*)(p + 16);
  v4f f3 = *(const v4f*)(p + 20);
  v16h a;
#pragma unroll
  for (int i = 0; i < 4; ++i) {
    a[i]      = (_Float16)f0[i];
    a[i + 4]  = (_Float16)f1[i];
    a[i + 8]  = (_Float16)f2[i];
    a[i + 12] = (_Float16)f3[i];
  }
  return a;
}

// Store C tile (f32 accum) with bias (+ optional ReLU) as f16 rows into LDS.
// C layout: lane l -> N = n0 + (l&15); VGPR r -> M = (l>>4)*8 + r.
DEVINL void store_c_relu_lds(_Float16* buf, int ld, int n0, v8f c,
                             const float* bias, int lane, bool relu) {
  int n  = n0 + (lane & 15);
  int mb = (lane >> 4) * 8;
  float bv = bias[n];
#pragma unroll
  for (int r = 0; r < 8; ++r) {
    float x = c[r] + bv;
    if (relu) x = x > 0.f ? x : 0.f;
    buf[(size_t)(mb + r) * ld + n] = (_Float16)x;
  }
}

// Generic 16xN GEMM N-tile: A (16 x 32*Ktiles) from LDS, W fragments from ws.
// Two interleaved accumulators break the WMMA->WMMA RAW chain.
DEVINL v8f gemm_tile_lds(const _Float16* A, int ldA, const _Float16* W,
                         int nt, int Ktiles, int lane) {
  v8f c0 = {}, c1 = {};
  int kt = 0;
  for (; kt + 1 < Ktiles; kt += 2) {
    v16h a0 = load_a_frag_lds(A, ldA, kt * 32, lane);
    v16h b0 = load_b_frag(W, nt * Ktiles + kt, lane);
    c0 = wmma_f16(a0, b0, c0);
    v16h a1 = load_a_frag_lds(A, ldA, (kt + 1) * 32, lane);
    v16h b1 = load_b_frag(W, nt * Ktiles + kt + 1, lane);
    c1 = wmma_f16(a1, b1, c1);
  }
  if (kt < Ktiles) {
    v16h a0 = load_a_frag_lds(A, ldA, kt * 32, lane);
    v16h b0 = load_b_frag(W, nt * Ktiles + kt, lane);
    c0 = wmma_f16(a0, b0, c0);
  }
  return c0 + c1;
}

// A fragment for token-projection GEMMs: the (16 rows x 8 tok x 32) feature
// block viewed as a 128x32 matrix; M-tile mt selects flat rows mt*16..mt*16+15.
// fb points at the 256-wide feature region inside the merged (ld=MLD) buffer.
DEVINL v16h load_a_frag_tok(const _Float16* fb, int mt, int lane) {
  int flat = mt * 16 + (lane & 15);
  int r = flat >> 3, t = flat & 7;
  int ks = (lane >> 4) * 8;
  const _Float16* p = fb + (size_t)r * MLD + t * 32 + ks;
  v8h lo = *(const v8h*)(p);
  v8h hi = *(const v8h*)(p + 16);
  v16h a;
#pragma unroll
  for (int i = 0; i < 8; ++i) { a[i] = lo[i]; a[i + 8] = hi[i]; }
  return a;
}

// Store projection C tile into a 128-row (ld=PLD) f16 scratch.
DEVINL void store_c_proj(_Float16* dst, int mt, int nt, v8f c, int lane) {
  int n  = nt * 16 + (lane & 15);
  int fb = mt * 16 + (lane >> 4) * 8;
#pragma unroll
  for (int r = 0; r < 8; ++r) dst[(size_t)(fb + r) * PLD + n] = (_Float16)c[r];
}

// ---------------------------------------------------------------------------
// Weight conversion: f32 (K x Nsrc row-major) -> f16 B-fragment order.
// frag f = nt*Ktiles + kt; element (l,j): K = kt*32 + (l>>4)*8 + (j&7) + (j>>3)*16,
//                                          N = nt*16 + (l&15).  Pad N with 0.
// ---------------------------------------------------------------------------
__global__ void convert_weights(const float* __restrict__ src,
                                _Float16* __restrict__ dst,
                                int K, int Nsrc, int Npad) {
  int gid = blockIdx.x * blockDim.x + threadIdx.x;
  int total = K * Npad;
  if (gid >= total) return;
  int Ktiles = K >> 5;
  int f   = gid >> 9;
  int rem = gid & 511;
  int l = rem >> 4;
  int j = rem & 15;
  int kt = f % Ktiles;
  int nt = f / Ktiles;
  int k = kt * 32 + (l >> 4) * 8 + (j & 7) + (j >> 3) * 16;
  int n = nt * 16 + (l & 15);
  float v = (n < Nsrc) ? src[(size_t)k * Nsrc + n] : 0.f;
  dst[gid] = (_Float16)v;
}

// ---------------------------------------------------------------------------
// Fused policy network: 4 waves per block cooperate on one 16-row batch tile,
// splitting every N-tile loop (nt = wave, wave+4, ...) and sharing one 62 KB
// LDS slice; split-barriers between layer stages.
// Block LDS (halfs): merged[16 x MLD] (feat0|attn0|feat1|attn1) + scr[15360]
//   scr reused as: E1/merge1 hidden (16 x HLD), q/k/v token buffers, merge2 out.
// ---------------------------------------------------------------------------
__global__ __launch_bounds__(32 * NWAVES)
void fused_policy(const float* __restrict__ joint,
                  const float* __restrict__ eb1, const float* __restrict__ eb2,
                  const float* __restrict__ mb1, const float* __restrict__ mb2,
                  const float* __restrict__ lb,
                  const _Float16* __restrict__ wsE1, const _Float16* __restrict__ wsE2,
                  const _Float16* __restrict__ wsQ,  const _Float16* __restrict__ wsK,
                  const _Float16* __restrict__ wsV,
                  const _Float16* __restrict__ wsM1, const _Float16* __restrict__ wsM2,
                  const _Float16* __restrict__ wsLG,
                  float* __restrict__ out) {
  extern __shared__ _Float16 smem[];
  const int lane = threadIdx.x & 31;
  const int wave = threadIdx.x >> 5;
  _Float16* merged = smem;                  // 16 x MLD (shared)
  _Float16* scr    = smem + MERGED_HALFS;   // shared scratch
  const int rowBase = blockIdx.x * 16;

  // ---------------- Siamese encoder (both halves) ----------------
  for (int half = 0; half < 2; ++half) {
    const float* obs = joint + (size_t)rowBase * 512 + half * 256;
    // Cache the 8 input A-fragments (16x256) in registers, reused over N.
    v16h ain[8];
#pragma unroll
    for (int kt = 0; kt < 8; ++kt)
      ain[kt] = load_a_frag_global_f32(obs, 512, kt * 32, lane);

    _Float16* hbuf = scr;                   // 16 x 512 hidden, ld HLD
#pragma unroll 1
    for (int nt = wave; nt < 32; nt += NWAVES) {   // E1: 16x256 @ 256x512
      if (nt + NWAVES < 32)
        prefetch_chunk(wsE1 + (size_t)(nt + NWAVES) * 8 * 512, lane, 128);
      v8f c0 = {}, c1 = {};
#pragma unroll
      for (int kt = 0; kt < 8; kt += 2) {
        c0 = wmma_f16(ain[kt],     load_b_frag(wsE1, nt * 8 + kt,     lane), c0);
        c1 = wmma_f16(ain[kt + 1], load_b_frag(wsE1, nt * 8 + kt + 1, lane), c1);
      }
      store_c_relu_lds(hbuf, HLD, nt * 16, c0 + c1, eb1, lane, true);
    }
    __syncthreads();

    _Float16* feat = merged + half * 512;   // feat0 @ col 0, feat1 @ col 512
#pragma unroll 1
    for (int nt = wave; nt < 16; nt += NWAVES) {   // E2: 16x512 @ 512x256
      if (nt + NWAVES < 16)
        prefetch_chunk(wsE2 + (size_t)(nt + NWAVES) * 16 * 512, lane, 256);
      v8f c = gemm_tile_lds(hbuf, HLD, wsE2, nt, 16, lane);
      store_c_relu_lds(feat, MLD, nt * 16, c, eb2, lane, true);
    }
    __syncthreads();   // feat complete; hbuf free for next half / attention
  }

  // ---------------- Cross attention (two directions) ----------------
  _Float16* qb = scr;                       // 128 x 32 each, ld PLD
  _Float16* kb = scr + 128 * PLD;
  _Float16* vb = scr + 2 * 128 * PLD;
#pragma unroll 1
  for (int dir = 0; dir < 2; ++dir) {
    const _Float16* srcFeat = merged + dir * 512;        // q side
    const _Float16* tgtFeat = merged + (1 - dir) * 512;  // k,v side
    // q/k/v projections via WMMA: (128x32) @ (32x32); mt split across waves.
#pragma unroll 1
    for (int mt = wave; mt < 8; mt += NWAVES) {
      v16h aq = load_a_frag_tok(srcFeat, mt, lane);
      v16h at = load_a_frag_tok(tgtFeat, mt, lane);
#pragma unroll
      for (int nt = 0; nt < 2; ++nt) {
        v8f cq = {}; cq = wmma_f16(aq, load_b_frag(wsQ, nt, lane), cq);
        store_c_proj(qb, mt, nt, cq, lane);
        v8f ck = {}; ck = wmma_f16(at, load_b_frag(wsK, nt, lane), ck);
        store_c_proj(kb, mt, nt, ck, lane);
        v8f cv = {}; cv = wmma_f16(at, load_b_frag(wsV, nt, lane), cv);
        store_c_proj(vb, mt, nt, cv, lane);
      }
    }
    __syncthreads();

    // Scalar 8x8 softmax-attention: 128 (row, q-token) work items = 128 threads.
    _Float16* aout = merged + 256 + dir * 512;
    {
      int w = threadIdx.x;
      int r = w >> 3, tq = w & 7;
      const _Float16* qv = qb + (size_t)(r * 8 + tq) * PLD;
      float s[8];
      float mx = -3.0e38f;
#pragma unroll
      for (int tk = 0; tk < 8; ++tk) {
        const _Float16* kv = kb + (size_t)(r * 8 + tk) * PLD;
        float d = 0.f;
#pragma unroll
        for (int e = 0; e < 32; ++e) d += (float)qv[e] * (float)kv[e];
        d *= 0.17677669529663687f;   // 1/sqrt(32)
        s[tk] = d;
        mx = fmaxf(mx, d);
      }
      float sum = 0.f;
#pragma unroll
      for (int tk = 0; tk < 8; ++tk) { s[tk] = __expf(s[tk] - mx); sum += s[tk]; }
      float inv = 1.f / sum;
#pragma unroll
      for (int e = 0; e < 32; ++e) {
        float o = 0.f;
#pragma unroll
        for (int tk = 0; tk < 8; ++tk)
          o += s[tk] * (float)vb[(size_t)(r * 8 + tk) * PLD + e];
        aout[(size_t)r * MLD + tq * 32 + e] = (_Float16)(o * inv);
      }
    }
    __syncthreads();   // attn written; q/k/v buffers free for next dir
  }

  // ---------------- Merge MLP ----------------
  _Float16* hbuf = scr;                     // 16 x 512, ld HLD
#pragma unroll 1
  for (int nt = wave; nt < 32; nt += NWAVES) {   // merge1: 16x1024 @ 1024x512
    if (nt + NWAVES < 32)
      prefetch_chunk(wsM1 + (size_t)(nt + NWAVES) * 32 * 512, lane, 512);
    v8f c = gemm_tile_lds(merged, MLD, wsM1, nt, 32, lane);
    store_c_relu_lds(hbuf, HLD, nt * 16, c, mb1, lane, true);
  }
  __syncthreads();

  _Float16* h2 = scr + 16 * HLD;            // 16 x 256, ld H2LD
#pragma unroll 1
  for (int nt = wave; nt < 16; nt += NWAVES) {   // merge2: 16x512 @ 512x256
    if (nt + NWAVES < 16)
      prefetch_chunk(wsM2 + (size_t)(nt + NWAVES) * 16 * 512, lane, 256);
    v8f c = gemm_tile_lds(hbuf, HLD, wsM2, nt, 16, lane);
    store_c_relu_lds(h2, H2LD, nt * 16, c, mb2, lane, true);
  }
  __syncthreads();

  // logits: 16x256 @ 256x16(padded) -> 16x10 f32 to global (wave 0 only)
  if (wave == 0) {
    v8f c = gemm_tile_lds(h2, H2LD, wsLG, 0, 8, lane);
    int n  = lane & 15;
    int mb = (lane >> 4) * 8;
    if (n < 10) {
      float bv = lb[n];
#pragma unroll
      for (int r = 0; r < 8; ++r)
        out[(size_t)(rowBase + mb + r) * 10 + n] = c[r] + bv;
    }
  }
}

// ---------------------------------------------------------------------------
// Workspace layout (in halfs) — every offset is a multiple of 512 (32B align).
// ---------------------------------------------------------------------------
#define WS_E1 0u            // 256*512  = 131072
#define WS_E2 131072u       // 512*256  = 131072
#define WS_Q  262144u       // 32*32    = 1024
#define WS_K  263168u
#define WS_V  264192u
#define WS_M1 265216u       // 1024*512 = 524288
#define WS_M2 789504u       // 512*256  = 131072
#define WS_LG 920576u       // 256*16   = 4096

extern "C" void kernel_launch(void* const* d_in, const int* in_sizes, int n_in,
                              void* d_out, int out_size, void* d_ws, size_t ws_size,
                              hipStream_t stream) {
  const float* joint = (const float*)d_in[0];
  const float* ew1   = (const float*)d_in[1];
  const float* eb1   = (const float*)d_in[2];
  const float* ew2   = (const float*)d_in[3];
  const float* eb2   = (const float*)d_in[4];
  const float* qw    = (const float*)d_in[5];
  const float* kw    = (const float*)d_in[6];
  const float* vw    = (const float*)d_in[7];
  const float* mw1   = (const float*)d_in[8];
  const float* mb1   = (const float*)d_in[9];
  const float* mw2   = (const float*)d_in[10];
  const float* mb2   = (const float*)d_in[11];
  const float* lw    = (const float*)d_in[12];
  const float* lb    = (const float*)d_in[13];
  float* out = (float*)d_out;
  _Float16* ws = (_Float16*)d_ws;

  auto conv = [&](const float* src, _Float16* dst, int K, int Nsrc, int Npad) {
    int total = K * Npad;
    convert_weights<<<(total + 255) / 256, 256, 0, stream>>>(src, dst, K, Nsrc, Npad);
  };
  conv(ew1, ws + WS_E1, 256, 512, 512);
  conv(ew2, ws + WS_E2, 512, 256, 256);
  conv(qw,  ws + WS_Q,  32, 32, 32);
  conv(kw,  ws + WS_K,  32, 32, 32);
  conv(vw,  ws + WS_V,  32, 32, 32);
  conv(mw1, ws + WS_M1, 1024, 512, 512);
  conv(mw2, ws + WS_M2, 512, 256, 256);
  conv(lw,  ws + WS_LG, 256, 10, 16);

  int rows    = in_sizes[0] / 512;   // B
  int nblocks = rows / 16;           // one 16-row tile per block (4 waves)
  size_t smem_bytes = (size_t)BLK_HALFS * sizeof(_Float16);  // ~62 KB
  fused_policy<<<nblocks, 32 * NWAVES, smem_bytes, stream>>>(
      joint, eb1, eb2, mb1, mb2, lb,
      ws + WS_E1, ws + WS_E2, ws + WS_Q, ws + WS_K, ws + WS_V,
      ws + WS_M1, ws + WS_M2, ws + WS_LG, out);
  (void)n_in; (void)out_size; (void)ws_size;
}
